// SCAttDec_13743895347280
// MI455X (gfx1250) — matrix operand order repaired and checked by reading, not compiled, not run on hardware
//
#include <hip/hip_runtime.h>
#include <stdint.h>

// Problem constants (match reference)
#define Bb   128
#define Hh   8
#define Mm   196
#define Dd   128
#define MIDn 64
#define DVv  128
#define MT   13            // ceil(196/16) M-tiles
#define Mpad 208           // MT*16, zero-padded stats arrays

typedef __attribute__((ext_vector_type(16))) __bf16 v16bf;
typedef __attribute__((ext_vector_type(8)))  float  v8f;
typedef __attribute__((__vector_size__(16))) int    vi4;   // b128 payload type
typedef __attribute__((address_space(1))) vi4* gvi4p;      // global (AS1)
typedef __attribute__((address_space(3))) vi4* lvi4p;      // LDS (AS3)

#if __has_builtin(__builtin_amdgcn_global_load_async_to_lds_b128) && \
    __has_builtin(__builtin_amdgcn_s_wait_asynccnt)
#define HAVE_ASYNC 1
#else
#define HAVE_ASYNC 0
#endif

__device__ __forceinline__ __bf16 f2bf(float f) { return (__bf16)f; }

__device__ __forceinline__ void wait_async() {
#if HAVE_ASYNC
  __builtin_amdgcn_s_wait_asynccnt(0);
#endif
}

__device__ __forceinline__ void async_b128(const float* gsrc, float* ldst) {
#if HAVE_ASYNC
  __builtin_amdgcn_global_load_async_to_lds_b128(
      (gvi4p)const_cast<float*>(gsrc), (lvi4p)ldst, 0, 0);
#else
  *(float4*)ldst = *(const float4*)gsrc;
#endif
}

__global__ __launch_bounds__(128)
void scatt_dec_fused(const float* __restrict__ query,
                     const float* __restrict__ key,
                     const float* __restrict__ att_mask,
                     const float* __restrict__ value1,
                     const float* __restrict__ value2,
                     const float* __restrict__ W_basic,
                     const float* __restrict__ b_basic,
                     const float* __restrict__ W_spatial,
                     const float* __restrict__ b_spatial,
                     const float* __restrict__ W_channel,
                     const float* __restrict__ b_channel,
                     float* __restrict__ out) {
  __shared__ __bf16 Wb[Dd][MIDn + 8];       // bf16 W_basic, padded stride
  __shared__ __bf16 At[16][Dd + 8];         // bf16 A tile (q*key), padded
  __shared__ float  key_tile[2][16][Dd];    // double-buffered raw key tile (async dst)
  __shared__ float  qv[Dd];
  __shared__ float  maskv[Mpad];            // zero-padded past Mm
  __shared__ float  alpha[Mpad];            // zero-padded past Mm
  __shared__ float  bb[MIDn];
  __shared__ float  Ws[MIDn];
  __shared__ float  pooled[MIDn];
  __shared__ float  wgt[Mm];
  __shared__ float  red[128];
  __shared__ float  sc_cnt, sc_bsp;

  const int tid  = threadIdx.x;
  const int bh   = blockIdx.x;              // 0 .. B*H-1
  const int bidx = bh / Hh;
  const int lane = tid & 31;
  const int wv   = tid >> 5;                // wave id == MID n-tile (0..3)
  const int g    = lane >> 4;               // lane half
  const int nl   = lane & 15;
  const int ecol = wv * 16 + nl;            // owned MID column (C/D layout N)

  // ---- kick async prefetch of key tile 0 --------------------------------
  {
    #pragma unroll
    for (int j = 0; j < 4; ++j) {
      int c   = tid + 128 * j;              // 16B-chunk id, 0..511
      int row = c >> 5;                     // 32 chunks per 512B row
      int m   = row;                        // tile 0
      int col = (c & 31) << 2;              // float index, step 4
      if (m < Mm) {
        async_b128(key + ((size_t)bh * Mm + m) * Dd + col,
                   &key_tile[0][row][col]);
      }
    }
  }

  // ---- Phase 0: stage constants into LDS (overlaps async copy) ----------
  for (int i = tid; i < Dd * MIDn; i += 128) {
    int d = i / MIDn, e = i % MIDn;
    Wb[d][e] = f2bf(W_basic[i]);
  }
  for (int i = tid; i < Dd; i += 128) qv[i] = query[(size_t)bh * Dd + i];
  for (int i = tid; i < Mpad; i += 128) {
    maskv[i] = (i < Mm) ? att_mask[(size_t)bidx * Mm + i] : 0.0f;
    alpha[i] = 0.0f;
  }
  for (int i = tid; i < MIDn; i += 128) {
    bb[i] = b_basic[i]; Ws[i] = W_spatial[i]; pooled[i] = 0.0f;
  }
  if (tid == 0) sc_bsp = b_spatial[0];
  __syncthreads();

  // mask count (denominator of masked mean pool)
  float c0 = 0.0f;
  for (int i = tid; i < Mm; i += 128) c0 += maskv[i];
  red[tid] = c0; __syncthreads();
  for (int s = 64; s > 0; s >>= 1) { if (tid < s) red[tid] += red[tid + s]; __syncthreads(); }
  if (tid == 0) sc_cnt = red[0];

  // ---- Preload B fragments (constant over M): 32x16 bf16 per k-step -----
  // B layout: elem e -> K=(e>=8?16:0)+g*8+(e&7); N = lane&15
  v16bf bfrag[4];
  #pragma unroll
  for (int kk = 0; kk < 4; ++kk) {
    v16bf t;
    #pragma unroll
    for (int e = 0; e < 16; ++e) {
      int K = kk * 32 + ((e >= 8) ? 16 : 0) + g * 8 + (e & 7);
      t[e] = Wb[K][ecol];
    }
    bfrag[kk] = t;
  }

  // ---- Main loop over 13 M-tiles: async-fed WMMA GEMM + fused stats -----
  float pool_acc = 0.0f;
  const float bias = bb[ecol];
  const float wsp  = Ws[ecol];

  for (int t = 0; t < MT; ++t) {
    wait_async();
    __syncthreads();                        // key_tile[t&1] ready; At free

    // cooperatively build bf16 A tile from staged key: At[r][d] = q[d]*key
    const int buf = t & 1;
    for (int i = tid; i < 16 * Dd; i += 128) {
      int r = i >> 7, d = i & (Dd - 1);
      int m = t * 16 + r;
      float kvf = (m < Mm) ? key_tile[buf][r][d] : 0.0f;
      At[r][d] = f2bf(qv[d] * kvf);
    }

    // prefetch next key tile into the other buffer (last read at tile t-1)
    if (t + 1 < MT) {
      const int nbuf = (t + 1) & 1;
      #pragma unroll
      for (int j = 0; j < 4; ++j) {
        int c   = tid + 128 * j;
        int row = c >> 5;
        int m   = (t + 1) * 16 + row;
        int col = (c & 31) << 2;
        if (m < Mm) {
          async_b128(key + ((size_t)bh * Mm + m) * Dd + col,
                     &key_tile[nbuf][row][col]);
        }
      }
    }
    __syncthreads();                        // At ready

    v8f c = {};
    #pragma unroll
    for (int kk = 0; kk < 4; ++kk) {
      // A layout: M = lane&15; elem e: v=e>>1,h=e&1 -> K=(v>=4?16:0)+g*8+2*(v&3)+h
      v16bf af;
      #pragma unroll
      for (int e = 0; e < 16; ++e) {
        int vv = e >> 1, hf = e & 1;
        int K  = kk * 32 + ((vv >= 4) ? 16 : 0) + g * 8 + 2 * (vv & 3) + hf;
        af[e] = At[nl][K];
      }
      c = __builtin_amdgcn_wmma_f32_16x16x32_bf16(
              false, af, false, bfrag[kk], (short)0, c, false, false);
    }

    // consume C fragment: h = relu(c + bias); fused pool + alpha partials.
    // mask/alpha are zero-padded to 208, so no validity branches needed.
    #pragma unroll
    for (int r = 0; r < 8; ++r) {
      int m    = t * 16 + r + g * 8;        // C/D layout: M = r + 8*(lane>=16)
      float hv = fmaxf(c[r] + bias, 0.0f);
      pool_acc += hv * maskv[m];
      float ap = hv * wsp;
      #pragma unroll
      for (int off = 1; off < 16; off <<= 1) ap += __shfl_xor(ap, off, 16);
      if (nl == 0) atomicAdd(&alpha[m], ap);
    }
  }
  atomicAdd(&pooled[ecol], pool_acc);
  __syncthreads();

  // ---- Masked softmax over alpha[0..195] --------------------------------
  float lmax = -1e30f;
  for (int m = tid; m < Mm; m += 128) {
    float a = (maskv[m] == 0.0f) ? -1.0e9f : (alpha[m] + sc_bsp);
    wgt[m] = a;
    lmax = fmaxf(lmax, a);
  }
  red[tid] = lmax; __syncthreads();
  for (int s = 64; s > 0; s >>= 1) { if (tid < s) red[tid] = fmaxf(red[tid], red[tid + s]); __syncthreads(); }
  float amax = red[0]; __syncthreads();

  float lsum = 0.0f;
  for (int m = tid; m < Mm; m += 128) {
    float e = __expf(wgt[m] - amax);
    wgt[m] = e;
    lsum += e;
  }
  red[tid] = lsum; __syncthreads();
  for (int s = 64; s > 0; s >>= 1) { if (tid < s) red[tid] += red[tid + s]; __syncthreads(); }
  float ssum = red[0]; __syncthreads();

  // ---- v2 = softmax . value2 ; chan = sigmoid(pool @ W_channel) ; fuse --
  const int v = tid;                        // 128 threads == DV columns
  float acc = 0.0f;
  const float* v2p = value2 + (size_t)bh * Mm * DVv + v;
  for (int m = 0; m < Mm; ++m) acc += wgt[m] * v2p[(size_t)m * DVv];
  float v2v = acc / ssum;

  float s = b_channel[v];
  float inv_cnt = 1.0f / sc_cnt;
  #pragma unroll 4
  for (int e = 0; e < MIDn; ++e) s += (pooled[e] * inv_cnt) * W_channel[e * DVv + v];
  float ch = 1.0f / (1.0f + __expf(-s));

  out[(size_t)bh * DVv + v] = value1[(size_t)bh * DVv + v] * v2v * ch;
}

extern "C" void kernel_launch(void* const* d_in, const int* in_sizes, int n_in,
                              void* d_out, int out_size, void* d_ws, size_t ws_size,
                              hipStream_t stream) {
  (void)in_sizes; (void)n_in; (void)out_size; (void)d_ws; (void)ws_size;
  const float* query     = (const float*)d_in[0];
  const float* key       = (const float*)d_in[1];
  const float* att_mask  = (const float*)d_in[2];
  const float* value1    = (const float*)d_in[3];
  const float* value2    = (const float*)d_in[4];
  const float* W_basic   = (const float*)d_in[5];
  const float* b_basic   = (const float*)d_in[6];
  const float* W_spatial = (const float*)d_in[7];
  const float* b_spatial = (const float*)d_in[8];
  const float* W_channel = (const float*)d_in[9];
  const float* b_channel = (const float*)d_in[10];
  float* out = (float*)d_out;

  dim3 grid(Bb * Hh), block(128);
  scatt_dec_fused<<<grid, block, 0, stream>>>(query, key, att_mask, value1, value2,
                                              W_basic, b_basic, W_spatial, b_spatial,
                                              W_channel, b_channel, out);
}
